// TCPR_40931038331405
// MI455X (gfx1250) — matrix operand order repaired and checked by reading, not compiled, not run on hardware
//
#include <hip/hip_runtime.h>
#include <hip/hip_bf16.h>
#include <math.h>

// Problem constants (from reference)
#define EP 8
#define NR 1024
#define DD 1024
#define NS 128
#define NB 100000
#define TOPK 15000
#define NCHUNK 250
#define CHUNK 400   // NCHUNK*CHUNK == NB

typedef __attribute__((ext_vector_type(16))) __bf16 v16bf;
typedef __attribute__((ext_vector_type(8)))  float  v8f;

__device__ __forceinline__ unsigned okey(float f) {
  unsigned u = __float_as_uint(f);
  return (u & 0x80000000u) ? ~u : (u | 0x80000000u);
}

// fixed-order block reduction (256 threads), deterministic
__device__ __forceinline__ float block_reduce_sum(float v, float* buf) {
  int t = threadIdx.x;
  buf[t] = v; __syncthreads();
  #pragma unroll
  for (int s = 128; s > 0; s >>= 1) {
    if (t < s) buf[t] += buf[t + s];
    __syncthreads();
  }
  float r = buf[0]; __syncthreads();
  return r;
}

// --- K1: q[e] = normalize(sum of normalized support rows); rows 8..15 zero-padded for WMMA A
__global__ __launch_bounds__(256) void k_support_q(const float* __restrict__ feat,
                                                   float* __restrict__ q) {
  __shared__ float buf[256];
  int e = blockIdx.x, t = threadIdx.x;
  if (e >= EP) {
    float4 z = make_float4(0.f, 0.f, 0.f, 0.f);
    ((float4*)(q + (size_t)e * DD))[t] = z;
    return;
  }
  float4 acc = make_float4(0.f, 0.f, 0.f, 0.f);
  for (int r = 0; r < NS; ++r) {
    float4 v = ((const float4*)(feat + ((size_t)e * NR + r) * DD))[t];
    float ssq = block_reduce_sum(v.x*v.x + v.y*v.y + v.z*v.z + v.w*v.w, buf);
    float inv = 1.0f / fmaxf(sqrtf(ssq), 1e-12f);
    acc.x += v.x*inv; acc.y += v.y*inv; acc.z += v.z*inv; acc.w += v.w*inv;
  }
  float ssq = block_reduce_sum(acc.x*acc.x + acc.y*acc.y + acc.z*acc.z + acc.w*acc.w, buf);
  float inv = 1.0f / fmaxf(sqrtf(ssq), 1e-12f);
  float4 o = make_float4(acc.x*inv, acc.y*inv, acc.z*inv, acc.w*inv);
  ((float4*)(q + (size_t)e * DD))[t] = o;
}

// --- K2: WMMA GEMM  sim[e][b] = q[e] . base[b] / ||base[b]||  (bf16 x bf16 -> f32 acc)
// One wave per 16 base rows; fused fp32 row-norm. simT stored b-major (8 floats/row).
__global__ __launch_bounds__(256) void k_gemm_sim(const float* __restrict__ base,
                                                  const float* __restrict__ q,
                                                  float* __restrict__ simT,
                                                  float* __restrict__ rnorm) {
  int gwave = (int)((blockIdx.x * 256u + threadIdx.x) >> 5);
  int lane  = threadIdx.x & 31;
  if (gwave >= NB / 16) return;                 // wave-uniform: EXEC stays all-1s
  int col   = lane & 15;
  int khalf = lane >> 4;
  const float* brow = base + (size_t)(gwave * 16 + col) * DD + khalf * 16; // B: contiguous 16 K
  const float* arow = q    + (size_t)col * DD + khalf * 16;                // A: contiguous 16 K
  v8f c = {};
  float ssq = 0.f;
  for (int k0 = 0; k0 < DD; k0 += 32) {
    float4 b0 = *(const float4*)(brow + k0);
    float4 b1 = *(const float4*)(brow + k0 + 4);
    float4 b2 = *(const float4*)(brow + k0 + 8);
    float4 b3 = *(const float4*)(brow + k0 + 12);
    float4 a0 = *(const float4*)(arow + k0);
    float4 a1 = *(const float4*)(arow + k0 + 4);
    float4 a2 = *(const float4*)(arow + k0 + 8);
    float4 a3 = *(const float4*)(arow + k0 + 12);
    ssq += b0.x*b0.x + b0.y*b0.y + b0.z*b0.z + b0.w*b0.w
         + b1.x*b1.x + b1.y*b1.y + b1.z*b1.z + b1.w*b1.w
         + b2.x*b2.x + b2.y*b2.y + b2.z*b2.z + b2.w*b2.w
         + b3.x*b3.x + b3.y*b3.y + b3.z*b3.z + b3.w*b3.w;
    v16bf A, B;
    A[0]=(__bf16)a0.x; A[1]=(__bf16)a0.y; A[2]=(__bf16)a0.z; A[3]=(__bf16)a0.w;
    A[4]=(__bf16)a1.x; A[5]=(__bf16)a1.y; A[6]=(__bf16)a1.z; A[7]=(__bf16)a1.w;
    A[8]=(__bf16)a2.x; A[9]=(__bf16)a2.y; A[10]=(__bf16)a2.z; A[11]=(__bf16)a2.w;
    A[12]=(__bf16)a3.x; A[13]=(__bf16)a3.y; A[14]=(__bf16)a3.z; A[15]=(__bf16)a3.w;
    B[0]=(__bf16)b0.x; B[1]=(__bf16)b0.y; B[2]=(__bf16)b0.z; B[3]=(__bf16)b0.w;
    B[4]=(__bf16)b1.x; B[5]=(__bf16)b1.y; B[6]=(__bf16)b1.z; B[7]=(__bf16)b1.w;
    B[8]=(__bf16)b2.x; B[9]=(__bf16)b2.y; B[10]=(__bf16)b2.z; B[11]=(__bf16)b2.w;
    B[12]=(__bf16)b3.x; B[13]=(__bf16)b3.y; B[14]=(__bf16)b3.z; B[15]=(__bf16)b3.w;
    c = __builtin_amdgcn_wmma_f32_16x16x32_bf16(false, A, false, B, (short)0, c, false, false);
  }
  ssq += __shfl_xor(ssq, 16);                   // combine lane n with n+16 (wave32)
  float rn = 1.0f / fmaxf(sqrtf(ssq), 1e-12f);
  if (lane < 16) {
    int b = gwave * 16 + col;
    rnorm[b] = rn;
    float4 s0 = make_float4(c[0]*rn, c[1]*rn, c[2]*rn, c[3]*rn);
    float4 s1 = make_float4(c[4]*rn, c[5]*rn, c[6]*rn, c[7]*rn);
    float4* outp = (float4*)(simT + (size_t)b * 8);
    outp[0] = s0; outp[1] = s1;
  }
}

// --- selection state/init
__global__ void k_init(unsigned* __restrict__ hist, unsigned* __restrict__ state) {
  int t = threadIdx.x;
  for (int i = t; i < 4 * 8 * 256; i += 256) hist[i] = 0;
  if (t < EP) { state[t] = 0u; state[8 + t] = TOPK; }
}

// --- radix-select histogram pass
__global__ __launch_bounds__(256) void k_hist(const float* __restrict__ simT,
                                              unsigned* __restrict__ hist,
                                              const unsigned* __restrict__ state, int pass) {
  __shared__ unsigned h[256];
  int e = blockIdx.y, t = threadIdx.x;
  h[t] = 0; __syncthreads();
  unsigned prefix = state[e];
  int shift = 24 - 8 * pass;
  unsigned mask = pass ? (0xFFFFFFFFu << (shift + 8)) : 0u;
  int stride = gridDim.x * 256;
  for (int r = blockIdx.x * 256 + t; r < NB; r += stride) {
    unsigned k = okey(simT[(size_t)r * 8 + e]);
    if ((k & mask) == prefix) atomicAdd(&h[(k >> shift) & 255u], 1u);
  }
  __syncthreads();
  if (h[t]) atomicAdd(&hist[((size_t)pass * 8 + e) * 256 + t], h[t]);
}

__global__ void k_scan(const unsigned* __restrict__ hist, unsigned* __restrict__ state, int pass) {
  int e = threadIdx.x;
  if (e >= EP) return;
  unsigned prefix = state[e], rem = state[8 + e];
  const unsigned* h = hist + ((size_t)pass * 8 + e) * 256;
  int shift = 24 - 8 * pass;
  unsigned cum = 0; int b = 0;
  for (b = 255; b > 0; --b) {
    unsigned c = h[b];
    if (cum + c >= rem) break;
    cum += c;
  }
  state[e]     = prefix | ((unsigned)b << shift);
  state[8 + e] = rem - cum;   // after pass 3: #equal-to-threshold keys to include
}

// --- count key==t_key per chunk (for deterministic, index-ordered tie inclusion)
__global__ __launch_bounds__(256) void k_eqcnt(const float* __restrict__ simT,
                                               const unsigned* __restrict__ state,
                                               unsigned* __restrict__ eqc) {
  __shared__ unsigned cnt;
  int chunk = blockIdx.x, e = blockIdx.y, t = threadIdx.x;
  if (t == 0) cnt = 0;
  __syncthreads();
  unsigned tk = state[e], local = 0;
  for (int r = chunk * CHUNK + t; r < (chunk + 1) * CHUNK; r += 256)
    if (okey(simT[(size_t)r * 8 + e]) == tk) local++;
  if (local) atomicAdd(&cnt, local);
  __syncthreads();
  if (t == 0) eqc[(size_t)e * NCHUNK + chunk] = cnt;
}

__global__ void k_eqprefix(unsigned* __restrict__ eqc) {
  int e = threadIdx.x;
  if (e >= EP) return;
  unsigned run = 0;
  for (int c = 0; c < NCHUNK; ++c) {
    unsigned v = eqc[(size_t)e * NCHUNK + c];
    eqc[(size_t)e * NCHUNK + c] = run;   // exclusive prefix in-place
    run += v;
  }
}

// --- K5: per-chunk partial approx = sum over selected rows of sqrt(sim)*rnorm*base_row (all 8 eps)
// Rows iterated sequentially inside the block -> deterministic tie handling & summation order.
__global__ __launch_bounds__(256) void k_accum(const float* __restrict__ base,
                                               const float* __restrict__ simT,
                                               const float* __restrict__ rnorm,
                                               const unsigned* __restrict__ state,
                                               const unsigned* __restrict__ eqc,
                                               float* __restrict__ partial) {
  int chunk = blockIdx.x, t = threadIdx.x;
  unsigned tk[EP], need[EP], cnt[EP];
  #pragma unroll
  for (int e = 0; e < EP; ++e) {
    tk[e] = state[e]; need[e] = state[8 + e];
    cnt[e] = eqc[(size_t)e * NCHUNK + chunk];
  }
  float4 acc[EP];
  #pragma unroll
  for (int e = 0; e < EP; ++e) acc[e] = make_float4(0.f, 0.f, 0.f, 0.f);
  int r0 = chunk * CHUNK;
  for (int r = r0; r < r0 + CHUNK; ++r) {
    const float* s = simT + (size_t)r * 8;   // broadcast reads
    float rn = rnorm[r];
    float w[EP]; bool any = false;
    #pragma unroll
    for (int e = 0; e < EP; ++e) {
      float sv = s[e];
      unsigned k = okey(sv);
      bool inc = false;
      if (k > tk[e]) inc = true;
      else if (k == tk[e]) { if (cnt[e] < need[e]) inc = true; cnt[e]++; }
      w[e] = inc ? sqrtf(sv) * rn : 0.0f;
      any |= inc;
    }
    if (any) {
      float4 v = ((const float4*)(base + (size_t)r * DD))[t];
      #pragma unroll
      for (int e = 0; e < EP; ++e) {
        acc[e].x += w[e]*v.x; acc[e].y += w[e]*v.y;
        acc[e].z += w[e]*v.z; acc[e].w += w[e]*v.w;
      }
    }
  }
  #pragma unroll
  for (int e = 0; e < EP; ++e)
    ((float4*)(partial + ((size_t)chunk * 8 + e) * DD))[t] = acc[e];
}

// --- K6: reduce chunk partials (fixed order) + normalize approx
__global__ __launch_bounds__(256) void k_reduce_approx(const float* __restrict__ partial,
                                                       float* __restrict__ approxN) {
  __shared__ float buf[256];
  int e = blockIdx.x, t = threadIdx.x;
  float4 s = make_float4(0.f, 0.f, 0.f, 0.f);
  for (int c = 0; c < NCHUNK; ++c) {
    float4 v = ((const float4*)(partial + ((size_t)c * 8 + e) * DD))[t];
    s.x += v.x; s.y += v.y; s.z += v.z; s.w += v.w;
  }
  float ssq = block_reduce_sum(s.x*s.x + s.y*s.y + s.z*s.z + s.w*s.w, buf);
  float inv = 1.0f / fmaxf(sqrtf(ssq), 1e-12f);
  ((float4*)(approxN + (size_t)e * DD))[t] =
      make_float4(s.x*inv, s.y*inv, s.z*inv, s.w*inv);
}

// --- K7: out = normalize(f_norm - (f_norm . a) * a), f normalized on the fly
__global__ __launch_bounds__(256) void k_project(const float* __restrict__ feat,
                                                 const float* __restrict__ approxN,
                                                 float* __restrict__ out) {
  __shared__ float buf[256];
  int e = blockIdx.x >> 10, row = blockIdx.x & 1023, t = threadIdx.x;
  float4 f = ((const float4*)(feat + ((size_t)e * NR + row) * DD))[t];
  float4 a = ((const float4*)(approxN + (size_t)e * DD))[t];
  float ssq = block_reduce_sum(f.x*f.x + f.y*f.y + f.z*f.z + f.w*f.w, buf);
  float dot = block_reduce_sum(f.x*a.x + f.y*a.y + f.z*a.z + f.w*a.w, buf);
  float inv = 1.0f / fmaxf(sqrtf(ssq), 1e-12f);
  float cosv = dot * inv;                       // ||f_norm|| == ||a|| == 1
  float4 r = make_float4(f.x*inv - cosv*a.x, f.y*inv - cosv*a.y,
                         f.z*inv - cosv*a.z, f.w*inv - cosv*a.w);
  float ssr = block_reduce_sum(r.x*r.x + r.y*r.y + r.z*r.z + r.w*r.w, buf);
  float ir = 1.0f / fmaxf(sqrtf(ssr), 1e-12f);
  ((float4*)(out + ((size_t)e * NR + row) * DD))[t] =
      make_float4(r.x*ir, r.y*ir, r.z*ir, r.w*ir);
}

extern "C" void kernel_launch(void* const* d_in, const int* in_sizes, int n_in,
                              void* d_out, int out_size, void* d_ws, size_t ws_size,
                              hipStream_t stream) {
  (void)in_sizes; (void)n_in; (void)out_size; (void)ws_size;
  const float* feat = (const float*)d_in[0];
  const float* base = (const float*)d_in[1];
  // d_in[2] (support_labels) only defines n_support == 128; values unused by reference math.
  float* out = (float*)d_out;

  char* ws = (char*)d_ws;
  size_t off = 0;
  auto take = [&](size_t bytes) { size_t o = off; off += (bytes + 255) & ~(size_t)255; return o; };
  float*    q       = (float*)   (ws + take((size_t)16 * DD * 4));
  float*    simT    = (float*)   (ws + take((size_t)NB * 8 * 4));
  float*    rnorm   = (float*)   (ws + take((size_t)NB * 4));
  unsigned* hist    = (unsigned*)(ws + take((size_t)4 * 8 * 256 * 4));
  unsigned* state   = (unsigned*)(ws + take((size_t)16 * 4));
  unsigned* eqc     = (unsigned*)(ws + take((size_t)EP * NCHUNK * 4));
  float*    partial = (float*)   (ws + take((size_t)NCHUNK * 8 * DD * 4));
  float*    approxN = (float*)   (ws + take((size_t)EP * DD * 4));

  k_support_q<<<dim3(16), dim3(256), 0, stream>>>(feat, q);
  k_gemm_sim<<<dim3((NB / 16 * 32 + 255) / 256), dim3(256), 0, stream>>>(base, q, simT, rnorm);
  k_init<<<dim3(1), dim3(256), 0, stream>>>(hist, state);
  for (int p = 0; p < 4; ++p) {
    k_hist<<<dim3(25, 8), dim3(256), 0, stream>>>(simT, hist, state, p);
    k_scan<<<dim3(1), dim3(32), 0, stream>>>(hist, state, p);
  }
  k_eqcnt<<<dim3(NCHUNK, 8), dim3(256), 0, stream>>>(simT, state, eqc);
  k_eqprefix<<<dim3(1), dim3(32), 0, stream>>>(eqc);
  k_accum<<<dim3(NCHUNK), dim3(256), 0, stream>>>(base, simT, rnorm, state, eqc, partial);
  k_reduce_approx<<<dim3(8), dim3(256), 0, stream>>>(partial, approxN);
  k_project<<<dim3(EP * NR), dim3(256), 0, stream>>>(feat, approxN, out);
}
